// EndOfTripDelay_8899172237732
// MI455X (gfx1250) — matrix operand order repaired and checked by reading, not compiled, not run on hardware
//
#include <hip/hip_runtime.h>
#include <hip/hip_bf16.h>

// ---------------------------------------------------------------------------
// EndOfTripDelay GNN for gfx1250 (MI455X, wave32, WMMA).
//
// Pipeline (matches JAX reference):
//   for layer in {0,1}:  (din,dout,K=din*dout) = (16,32,512) / (32,32,1024)
//     h = leaky(edge_attr @ w1 + b1)                 [E,K]   (fused, LDS, f16)
//     W = h @ w2 + b2  -> msg = x[src] . W           (fused WMMA, f32 accum)
//     sums[dst] += msg (atomics);  agg = sums/max(cnt,1)
//     x = BN(agg + x@root + bias) (+leaky on layer 0)
//   g = masked segment-mean by graph; h = [g, time_emb]
//   MLP 36->64->32->1 with train-mode BN + leaky on hidden layers
// ---------------------------------------------------------------------------

typedef __attribute__((ext_vector_type(16))) _Float16 v16h;
typedef __attribute__((ext_vector_type(8)))  float    v8f;

__device__ __forceinline__ float leakyf(float x) { return x >= 0.f ? x : 0.01f * x; }

__device__ __forceinline__ void atomAddGlobal(float* p, float v) {
  (void)__hip_atomic_fetch_add(p, v, __ATOMIC_RELAXED, __HIP_MEMORY_SCOPE_AGENT);
}
__device__ __forceinline__ void atomAddShared(float* p, float v) {
  (void)__hip_atomic_fetch_add(p, v, __ATOMIC_RELAXED, __HIP_MEMORY_SCOPE_WORKGROUP);
}

// ---------------------------------------------------------------------------
// Pre-pack w2 (f32, [K,K] row-major: w2[k][col]) into f16 WMMA-B tile order:
//   out[((nt*KB + kk)*32 + lane)*16 + m] = w2[kk*32 + (m&7) + (m>=8?16:0) + (lane>=16?8:0)][nt*16 + (lane&15)]
// so each lane's 16 halves for a (nt,kk) tile are 32B-contiguous.
// ---------------------------------------------------------------------------
__global__ void pack_w2_kernel(const float* __restrict__ w2,
                               _Float16* __restrict__ out, int K) {
  int t = blockIdx.x * blockDim.x + threadIdx.x;
  if (t >= K * K) return;
  int KB   = K >> 5;
  int m    = t & 15;
  int lane = (t >> 4) & 31;
  int q    = t >> 9;
  int kk   = q % KB;
  int nt   = q / KB;
  int kg   = kk * 32 + (m & 7) + ((m >= 8) ? 16 : 0) + ((lane >= 16) ? 8 : 0);
  int col  = nt * 16 + (lane & 15);
  out[t] = (_Float16)w2[(size_t)kg * K + col];
}

// ---------------------------------------------------------------------------
// Per-dst edge counts (same for both layers).
// ---------------------------------------------------------------------------
__global__ void count_kernel(const long long* __restrict__ dst, float* cnt, int E) {
  int t = blockIdx.x * blockDim.x + threadIdx.x;
  if (t < E) atomAddGlobal(&cnt[(int)dst[t]], 1.f);
}

// ---------------------------------------------------------------------------
// Fused NNConv message kernel. One wave (32 threads) owns 16 edges.
//   Phase 1: h[16,K] = leaky(edge_attr@w1+b1) computed into LDS, pre-swizzled
//            into WMMA A layout (f16).
//   Phase 2: for each 16-wide column tile nt of W = h@w2+b2:
//            accumulate C = A x B over K with v_wmma_f32_16x16x32_f16,
//            then fold into msg[e][o] += x_src[e][i]*(C+b2), where the whole
//            tile shares one input channel i = nt/2 (dout==32).
//   Phase 3: scatter msg into sums[dst] with global f32 atomics.
// ---------------------------------------------------------------------------
template <int DIN>
__global__ __launch_bounds__(32)
void conv_fused_kernel(const float* __restrict__ xin,
                       const long long* __restrict__ src,
                       const long long* __restrict__ dst,
                       const float* __restrict__ ea,
                       const float* __restrict__ w1,
                       const float* __restrict__ b1,
                       const _Float16* __restrict__ w2h,
                       const float* __restrict__ b2,
                       float* __restrict__ sums,
                       int E) {
  constexpr int K  = DIN * 32;   // din*dout
  constexpr int KB = K / 32;     // WMMA k-steps
  constexpr int NT = K / 16;     // 16-wide output column tiles

  __shared__ _Float16 hbuf[K * 16];       // [KB][32 lanes][16 halves] A-layout
  __shared__ float    xs_lds[16 * DIN];
  __shared__ float    ea_lds[16 * 8];
  __shared__ int      src_lds[16];
  __shared__ int      dst_lds[16];

  const int lane  = threadIdx.x;
  const int row   = lane & 15;
  const int hi    = lane >> 4;            // 0 or 1
  const int ebase = blockIdx.x * 16;

  if (lane < 16) {
    int e = ebase + lane;
    bool v = e < E;
    src_lds[lane] = v ? (int)src[e] : 0;
    dst_lds[lane] = v ? (int)dst[e] : -1;
  }
  for (int idx = lane; idx < 16 * 8; idx += 32) {
    int e = ebase + (idx >> 3);
    ea_lds[idx] = (e < E) ? ea[(size_t)e * 8 + (idx & 7)] : 0.f;
  }
  __syncthreads();
  for (int idx = lane; idx < 16 * DIN; idx += 32) {
    int el = idx / DIN;
    xs_lds[idx] = xin[(size_t)src_lds[el] * DIN + (idx % DIN)];
  }

  // ---- Phase 1: edge MLP -> h in WMMA A layout ----
  float ea_r[8];
#pragma unroll
  for (int j = 0; j < 8; ++j) ea_r[j] = ea_lds[row * 8 + j];

  for (int kb = 0; kb < KB; ++kb) {
#pragma unroll
    for (int m = 0; m < 16; ++m) {
      // A-layout K index for this lane/element
      int kg = kb * 32 + (m & 7) + ((m >= 8) ? 16 : 0) + (hi ? 8 : 0);
      float v = b1[kg];
#pragma unroll
      for (int j = 0; j < 8; ++j) v = fmaf(ea_r[j], w1[j * K + kg], v);
      hbuf[((kb * 32 + lane) << 4) + m] = (_Float16)leakyf(v);
    }
  }
  __syncthreads();

  // ---- Phase 2: W = h@w2+b2 tiles via WMMA, folded into msg ----
  float msg[2][8];
#pragma unroll
  for (int ob = 0; ob < 2; ++ob)
#pragma unroll
    for (int r = 0; r < 8; ++r) msg[ob][r] = 0.f;

  for (int nt = 0; nt < NT; ++nt) {
    const _Float16* bp = w2h + ((size_t)nt * KB * 32 + lane) * 16;
    if (nt + 1 < NT)
      __builtin_prefetch((const void*)(bp + (size_t)KB * 512), 0, 1);
    v8f acc = {};
    for (int kk = 0; kk < KB; ++kk) {
      v16h a = *(const v16h*)(hbuf + ((kk * 32 + lane) << 4));
      v16h b = *(const v16h*)(bp + ((size_t)kk << 9));  // kk*32 lanes*16 halves
      acc = __builtin_amdgcn_wmma_f32_16x16x32_f16(
          /*neg_a=*/false, a, /*neg_b=*/false, b,
          /*c_mod=*/(short)0, acc, /*reuse_a=*/false, /*reuse_b=*/false);
    }
    float b2v = b2[nt * 16 + row];   // whole lane column shares one W column
    int   i   = nt >> 1;             // input channel of this tile (dout==32)
    int   ob  = nt & 1;              // output-channel half
#pragma unroll
    for (int r = 0; r < 8; ++r) {
      float xv = xs_lds[(r + (hi ? 8 : 0)) * DIN + i];
      msg[ob][r] += (acc[r] + b2v) * xv;
    }
  }

  // ---- Phase 3: scatter-add messages ----
#pragma unroll
  for (int ob = 0; ob < 2; ++ob)
#pragma unroll
    for (int r = 0; r < 8; ++r) {
      int el = r + (hi ? 8 : 0);
      int d  = dst_lds[el];
      if (d >= 0) atomAddGlobal(&sums[(size_t)d * 32 + ob * 16 + row], msg[ob][r]);
    }
}

// ---------------------------------------------------------------------------
// Node update: x_new = sums/max(cnt,1) + x@root + bias, plus BN batch stats.
// 256 threads = 8 nodes x 32 channels.
// ---------------------------------------------------------------------------
template <int DIN>
__global__ __launch_bounds__(256)
void node_update_kernel(const float* __restrict__ sums, const float* __restrict__ cnt,
                        const float* __restrict__ xin, const float* __restrict__ root,
                        const float* __restrict__ bias, float* __restrict__ xout,
                        float* __restrict__ stats, int N) {
  __shared__ float ls[32], lq[32];
  int t = threadIdx.x;
  if (t < 32) { ls[t] = 0.f; lq[t] = 0.f; }
  __syncthreads();
  int n = blockIdx.x * 8 + (t >> 5);
  int c = t & 31;
  if (n < N) {
    float v = sums[(size_t)n * 32 + c] / fmaxf(cnt[n], 1.f);
    float r = bias[c];
    for (int i = 0; i < DIN; ++i)
      r = fmaf(xin[(size_t)n * DIN + i], root[i * 32 + c], r);
    v += r;
    xout[(size_t)n * 32 + c] = v;
    atomAddShared(&ls[c], v);
    atomAddShared(&lq[c], v * v);
  }
  __syncthreads();
  if (t < 32) {
    atomAddGlobal(&stats[t], ls[t]);
    atomAddGlobal(&stats[32 + t], lq[t]);
  }
}

__global__ void bn_apply_kernel(float* __restrict__ x, const float* __restrict__ stats,
                                const float* __restrict__ gamma, const float* __restrict__ beta,
                                int N, int do_leaky) {
  int idx = blockIdx.x * blockDim.x + threadIdx.x;
  if (idx >= N * 32) return;
  int   c   = idx & 31;
  float nf  = (float)N;
  float mu  = stats[c] / nf;
  float var = stats[32 + c] / nf - mu * mu;
  float v   = gamma[c] * (x[idx] - mu) * rsqrtf(var + 1e-5f) + beta[c];
  x[idx] = do_leaky ? leakyf(v) : v;
}

// ---------------------------------------------------------------------------
// Masked per-graph mean pooling (accumulate phase).
// ---------------------------------------------------------------------------
__global__ void pool_kernel(const float* __restrict__ x, const long long* __restrict__ batch,
                            const unsigned char* __restrict__ mask,
                            float* __restrict__ gsum, float* __restrict__ gcnt, int N) {
  int idx = blockIdx.x * blockDim.x + threadIdx.x;
  if (idx >= N * 32) return;
  int n = idx >> 5, c = idx & 31;
  if (!mask[n]) return;
  int b = (int)batch[n];
  atomAddGlobal(&gsum[b * 32 + c], x[idx]);
  if (c == 0) atomAddGlobal(&gcnt[b], 1.f);
}

// ---------------------------------------------------------------------------
// Whole MLP head in one block (G<=64 graphs): 36->64->32->1 with train-mode BN.
// ---------------------------------------------------------------------------
__global__ __launch_bounds__(256)
void mlp_kernel(const float* __restrict__ gsum, const float* __restrict__ gcnt,
                const float* __restrict__ te,
                const float* __restrict__ W0, const float* __restrict__ B0,
                const float* __restrict__ G0, const float* __restrict__ BE0,
                const float* __restrict__ W1, const float* __restrict__ B1,
                const float* __restrict__ G1, const float* __restrict__ BE1,
                const float* __restrict__ W2, const float* __restrict__ B2,
                float* __restrict__ out, int G) {
  __shared__ float H0[64 * 36];
  __shared__ float H1[64 * 64];
  __shared__ float H2[64 * 32];
  __shared__ float mu[64], rs[64];
  int t = threadIdx.x;

  for (int idx = t; idx < G * 36; idx += 256) {
    int r = idx / 36, c = idx % 36;
    H0[idx] = (c < 32) ? gsum[r * 32 + c] / fmaxf(gcnt[r], 1.f) : te[r * 4 + (c - 32)];
  }
  __syncthreads();

  for (int idx = t; idx < G * 64; idx += 256) {
    int r = idx / 64, o = idx % 64;
    float a = B0[o];
    for (int i = 0; i < 36; ++i) a = fmaf(H0[r * 36 + i], W0[i * 64 + o], a);
    H1[idx] = a;
  }
  __syncthreads();
  if (t < 64) {
    float s = 0.f, q = 0.f;
    for (int r = 0; r < G; ++r) { float v = H1[r * 64 + t]; s += v; q += v * v; }
    float m = s / (float)G, var = q / (float)G - m * m;
    mu[t] = m; rs[t] = rsqrtf(var + 1e-5f);
  }
  __syncthreads();
  for (int idx = t; idx < G * 64; idx += 256) {
    int o = idx % 64;
    H1[idx] = leakyf((H1[idx] - mu[o]) * rs[o] * G0[o] + BE0[o]);
  }
  __syncthreads();

  for (int idx = t; idx < G * 32; idx += 256) {
    int r = idx / 32, o = idx % 32;
    float a = B1[o];
    for (int i = 0; i < 64; ++i) a = fmaf(H1[r * 64 + i], W1[i * 32 + o], a);
    H2[idx] = a;
  }
  __syncthreads();
  if (t < 32) {
    float s = 0.f, q = 0.f;
    for (int r = 0; r < G; ++r) { float v = H2[r * 32 + t]; s += v; q += v * v; }
    float m = s / (float)G, var = q / (float)G - m * m;
    mu[t] = m; rs[t] = rsqrtf(var + 1e-5f);
  }
  __syncthreads();
  for (int idx = t; idx < G * 32; idx += 256) {
    int o = idx % 32;
    H2[idx] = leakyf((H2[idx] - mu[o]) * rs[o] * G1[o] + BE1[o]);
  }
  __syncthreads();

  if (t < G) {
    float a = B2[0];
    for (int i = 0; i < 32; ++i) a = fmaf(H2[t * 32 + i], W2[i], a);
    out[t] = a;
  }
}

// ---------------------------------------------------------------------------
// Host side
// ---------------------------------------------------------------------------
extern "C" void kernel_launch(void* const* d_in, const int* in_sizes, int n_in,
                              void* d_out, int out_size, void* d_ws, size_t ws_size,
                              hipStream_t stream) {
  (void)ws_size; (void)out_size;
  enum { BN0_BETA, BN0_GAMMA, BN1_BETA, BN1_GAMMA,
         C0_B1, C0_B2, C0_BIAS, C0_ROOT, C0_W1, C0_W2,
         C1_B1, C1_B2, C1_BIAS, C1_ROOT, C1_W1, C1_W2,
         M0_W, M0_B, M0_BETA, M0_GAMMA,
         M1_W, M1_B, M1_BETA, M1_GAMMA,
         M2_W, M2_B, NPAR };

  // Resolve input ordering (three plausible pytree flattenings), using sizes.
  int idxX = 0, idxEI = 1, idxEA = 2, idxTM = 3, idxB = 4, idxTE = 5;
  int pmap[NPAR];
  if (n_in >= 33 && in_sizes[0] == 50000) {
    // fully key-sorted pytree: batch, edge_attr, edge_index, num_graphs, params..., time_emb, trip_mask, x
    idxB = 0; idxEA = 1; idxEI = 2; idxTE = 30; idxTM = 31; idxX = 32;
    for (int i = 0; i < NPAR; ++i) pmap[i] = 4 + i;
  } else if (n_in >= 33 && in_sizes[7] == 4096) {
    // insertion order everywhere: convs(w1,b1,w2,b2,root,bias)x2, bns(gamma,beta)x2, mlp(W,b,gamma,beta)
    static const int mc[NPAR] = {20, 19, 22, 21,
                                 8, 10, 12, 11, 7, 9,
                                 14, 16, 18, 17, 13, 15,
                                 23, 24, 26, 25,
                                 27, 28, 30, 29,
                                 31, 32};
    for (int i = 0; i < NPAR; ++i) pmap[i] = mc[i];
  } else {
    // top-level insertion order, params flattened with sorted keys (jax pytree)
    for (int i = 0; i < NPAR; ++i) pmap[i] = 7 + i;
  }

  const float*     x    = (const float*)d_in[idxX];
  const long long* eidx = (const long long*)d_in[idxEI];
  const float*     ea   = (const float*)d_in[idxEA];
  const unsigned char* mask = (const unsigned char*)d_in[idxTM];
  const long long* batch = (const long long*)d_in[idxB];
  const float*     te   = (const float*)d_in[idxTE];
  const int N = in_sizes[idxX] / 16;
  const int E = in_sizes[idxEA] / 8;
  const int G = in_sizes[idxTE] / 4;
  const long long* src = eidx;
  const long long* dst = eidx + E;

  auto P = [&](int i) { return (const float*)d_in[pmap[i]]; };

  // workspace carve-out (256B aligned slices)
  char* wsb = (char*)d_ws;
  size_t off = 0;
  auto carve = [&](size_t bytes) {
    void* p = wsb + off;
    off = (off + bytes + 255) & ~(size_t)255;
    return p;
  };
  _Float16* w2h0 = (_Float16*)carve((size_t)512 * 512 * 2);
  _Float16* w2h1 = (_Float16*)carve((size_t)1024 * 1024 * 2);
  float* xbuf0 = (float*)carve((size_t)N * 32 * 4);
  float* xbuf1 = (float*)carve((size_t)N * 32 * 4);
  float* sums  = (float*)carve((size_t)N * 32 * 4);
  float* cnt   = (float*)carve((size_t)N * 4);
  float* stats = (float*)carve(64 * 4);
  float* gsum  = (float*)carve((size_t)G * 32 * 4);
  float* gcnt  = (float*)carve((size_t)G * 4);

  const int eblocks = (E + 15) / 16;
  const int nblocks = (N + 7) / 8;
  const int elem    = (N * 32 + 255) / 256;

  // one-time packs + degree counts
  pack_w2_kernel<<<(512 * 512 + 255) / 256, 256, 0, stream>>>(P(C0_W2), w2h0, 512);
  pack_w2_kernel<<<(1024 * 1024 + 255) / 256, 256, 0, stream>>>(P(C1_W2), w2h1, 1024);
  hipMemsetAsync(cnt, 0, (size_t)N * 4, stream);
  count_kernel<<<(E + 255) / 256, 256, 0, stream>>>(dst, cnt, E);

  // ---- conv layer 0 (din=16) ----
  hipMemsetAsync(sums, 0, (size_t)N * 32 * 4, stream);
  hipMemsetAsync(stats, 0, 64 * 4, stream);
  conv_fused_kernel<16><<<eblocks, 32, 0, stream>>>(
      x, src, dst, ea, P(C0_W1), P(C0_B1), w2h0, P(C0_B2), sums, E);
  node_update_kernel<16><<<nblocks, 256, 0, stream>>>(
      sums, cnt, x, P(C0_ROOT), P(C0_BIAS), xbuf0, stats, N);
  bn_apply_kernel<<<elem, 256, 0, stream>>>(xbuf0, stats, P(BN0_GAMMA), P(BN0_BETA), N, 1);

  // ---- conv layer 1 (din=32) ----
  hipMemsetAsync(sums, 0, (size_t)N * 32 * 4, stream);
  hipMemsetAsync(stats, 0, 64 * 4, stream);
  conv_fused_kernel<32><<<eblocks, 32, 0, stream>>>(
      xbuf0, src, dst, ea, P(C1_W1), P(C1_B1), w2h1, P(C1_B2), sums, E);
  node_update_kernel<32><<<nblocks, 256, 0, stream>>>(
      sums, cnt, xbuf0, P(C1_ROOT), P(C1_BIAS), xbuf1, stats, N);
  bn_apply_kernel<<<elem, 256, 0, stream>>>(xbuf1, stats, P(BN1_GAMMA), P(BN1_BETA), N, 0);

  // ---- pool + MLP head ----
  hipMemsetAsync(gsum, 0, (size_t)G * 32 * 4, stream);
  hipMemsetAsync(gcnt, 0, (size_t)G * 4, stream);
  pool_kernel<<<elem, 256, 0, stream>>>(xbuf1, batch, mask, gsum, gcnt, N);
  mlp_kernel<<<1, 256, 0, stream>>>(gsum, gcnt, te,
                                    P(M0_W), P(M0_B), P(M0_GAMMA), P(M0_BETA),
                                    P(M1_W), P(M1_B), P(M1_GAMMA), P(M1_BETA),
                                    P(M2_W), P(M2_B), (float*)d_out, G);
}
